// InteractionNet_33956011442661
// MI455X (gfx1250) — compile-verified, matrix operand order checked by reading, not run on hardware
//
#include <hip/hip_runtime.h>

// ---------------------------------------------------------------------------
// CDNA5 / gfx1250 InteractionNet (NRI) forward pass.
// GEMMs: v_wmma_f32_16x16x32_bf16, double-buffered LDS tiles staged via
// GLOBAL_LOAD_ASYNC_TO_LDS_B128 when the toolchain exposes it.
// Weights are pre-transposed to (N,K) bf16 once per call so ALL tile staging
// is contiguous 16-byte copies (async-to-LDS friendly, no in-kernel transpose).
// ---------------------------------------------------------------------------

typedef __attribute__((ext_vector_type(16))) __bf16 v16bf;
typedef __attribute__((ext_vector_type(8)))  __bf16 v8bf;
typedef __attribute__((ext_vector_type(8)))  float  v8f;
typedef __attribute__((ext_vector_type(4)))  int    v4i;

#define Bsz   128
#define Nn    20
#define Ff    64
#define Hh    512
#define Ee    380           // N*(N-1)
#define Ll    40            // N*T
#define KTOT  194560        // E*H
#define NOUT  100

#if defined(__AMDGCN__) && __has_builtin(__builtin_amdgcn_global_load_async_to_lds_b128)
#define HAVE_ASYNC_LDS 1
#else
#define HAVE_ASYNC_LDS 0
#endif

__device__ __forceinline__ void async_b128(const void* g, void* l) {
#if HAVE_ASYNC_LDS
  __builtin_amdgcn_global_load_async_to_lds_b128(
      (__attribute__((address_space(1))) v4i*)(void*)g,
      (__attribute__((address_space(3))) v4i*)l, 0, 0);
#else
  *(uint4*)l = *(const uint4*)g;
#endif
}
__device__ __forceinline__ void wait_async_all() {
#if HAVE_ASYNC_LDS
#if __has_builtin(__builtin_amdgcn_s_wait_asynccnt)
  __builtin_amdgcn_s_wait_asynccnt(0);
#else
  asm volatile("s_wait_asynccnt 0x0" ::: "memory");
#endif
#endif
}

__device__ __forceinline__ __bf16 f2bf(float f) {
  unsigned u = __builtin_bit_cast(unsigned, f);
  unsigned r = u + 0x7FFFu + ((u >> 16) & 1u);   // round-to-nearest-even
  return __builtin_bit_cast(__bf16, (unsigned short)(r >> 16));
}
__device__ __forceinline__ float bf2f(__bf16 b) {
  unsigned u = ((unsigned)__builtin_bit_cast(unsigned short, b)) << 16;
  return __builtin_bit_cast(float, u);
}
__device__ __forceinline__ v16bf cat8(v8bf lo, v8bf hi) {
  return __builtin_shufflevector(lo, hi, 0,1,2,3,4,5,6,7,8,9,10,11,12,13,14,15);
}
__device__ __forceinline__ float elu(float v) {
  return v > 0.f ? v : (__expf(v) - 1.f);
}

// ---------------------------------------------------------------------------
// Elementwise helpers
// ---------------------------------------------------------------------------
__global__ void k_f32_to_bf16(const float* __restrict__ in, __bf16* __restrict__ out, long n) {
  long i = (long)blockIdx.x * blockDim.x + threadIdx.x;
  if (i < n) out[i] = f2bf(in[i]);
}

// weights: in (K,N) f32 -> out (N,K) bf16 (one-time; reads coalesced)
__global__ void k_cvt_transpose(const float* __restrict__ in, __bf16* __restrict__ out,
                                long K, int N) {
  long i = (long)blockIdx.x * blockDim.x + threadIdx.x;
  if (i >= K * N) return;
  long k = i / N; int n = (int)(i - k * N);
  out[(long)n * K + k] = f2bf(in[i]);
}

__global__ void k_zero_f32(float* p, int n) {
  int i = blockIdx.x * blockDim.x + threadIdx.x;
  if (i < n) p[i] = 0.f;
}

__global__ void k_out_init(float* __restrict__ out, const float* __restrict__ bias) {
  int i = blockIdx.x * blockDim.x + threadIdx.x;
  if (i < Bsz * NOUT) out[i] = bias[i % NOUT];
}

__global__ void k_argmax_rows(const float* __restrict__ m, int* __restrict__ idx,
                              int rows, int cols) {
  int r = blockIdx.x * blockDim.x + threadIdx.x;
  if (r >= rows) return;
  float best = m[(long)r * cols]; int bi = 0;
  for (int c = 1; c < cols; ++c) {
    float v = m[(long)r * cols + c];
    if (v > best) { best = v; bi = c; }
  }
  idx[r] = bi;
}

__global__ void k_build_reclist(const int* __restrict__ ridx, int* __restrict__ recList) {
  int n = threadIdx.x;
  if (n >= Nn) return;
  int cnt = 0;
  for (int e = 0; e < Ee; ++e)
    if (ridx[e] == n) recList[n * (Nn - 1) + (cnt++)] = e;
}

// ---------------------------------------------------------------------------
// Main bf16 WMMA GEMM:  C(M,N) = ELU(A(M,K) @ W + bias),  W given as WT (N,K).
// BM=128, BN=128, BK=32; 256 threads = 8 wave32s.
// Wave w owns M-rows [w*16, w*16+16) x 8 N-tiles; double-buffered async tiles.
// ---------------------------------------------------------------------------
#define LDA 40   // bf16 elems per LDS row (32 + 8 pad; 80B keeps 16B align, no bank repeat)

template<bool OUT_BF16>
__global__ __launch_bounds__(256)
void k_gemm_bf16(const __bf16* __restrict__ A, const __bf16* __restrict__ WT,
                 const float* __restrict__ bias, void* __restrict__ Cout,
                 int M, int K, int N) {
  __shared__ __bf16 lA[2][128 * LDA];
  __shared__ __bf16 lW[2][128 * LDA];

  const int tid  = threadIdx.x;
  const int lane = tid & 31;
  const int wave = tid >> 5;
  const long rowBase = (long)blockIdx.x * 128;
  const long colBase = (long)blockIdx.y * 128;

  v8f acc[8] = {};
  const int mrow = wave * 16 + (lane & 15);
  const int k0   = (lane >> 4) * 8;      // ISA 16-bit A layout: lanes 16-31 start at K=8

  auto stage = [&](int buf, int kb) {
#pragma unroll
    for (int i = 0; i < 2; ++i) {
      int idx = tid + i * 256;           // 512 uint4 each for A and WT tiles
      int r = idx >> 2, c = (idx & 3) * 8;
      async_b128(A  + (rowBase + r) * (long)K + kb + c, &lA[buf][r * LDA + c]);
      async_b128(WT + (colBase + r) * (long)K + kb + c, &lW[buf][r * LDA + c]);
    }
  };

  const int nIter = K >> 5;
  stage(0, 0);
  for (int it = 0; it < nIter; ++it) {
    wait_async_all();
    __syncthreads();
    if (it + 1 < nIter) stage((it + 1) & 1, (it + 1) << 5);

    const __bf16* cA = lA[it & 1];
    const __bf16* cW = lW[it & 1];
    v16bf a = cat8(*(const v8bf*)&cA[mrow * LDA + k0],
                   *(const v8bf*)&cA[mrow * LDA + k0 + 16]);
#pragma unroll
    for (int j = 0; j < 8; ++j) {
      int n = j * 16 + (lane & 15);
      v16bf b = cat8(*(const v8bf*)&cW[n * LDA + k0],
                     *(const v8bf*)&cW[n * LDA + k0 + 16]);
      acc[j] = __builtin_amdgcn_wmma_f32_16x16x32_bf16(
          false, a, false, b, (short)0, acc[j], false, false);
    }
  }

  // ---- epilogue: bias + ELU ----
#pragma unroll
  for (int j = 0; j < 8; ++j) {
#pragma unroll
    for (int i = 0; i < 8; ++i) {
      long m = rowBase + wave * 16 + (lane >> 4) * 8 + i;
      long n = colBase + j * 16 + (lane & 15);
      float v = elu(acc[j][i] + bias[n]);
      if (OUT_BF16) ((__bf16*)Cout)[m * N + n] = f2bf(v);
      else          ((float*) Cout)[m * N + n] = v;
    }
  }
}

// ---------------------------------------------------------------------------
// Final fc: out(128,100) += A(128,KTOT) @ W, W pre-transposed to (100,KTOT).
// Split-K over blocks, N padded to 112 (7 tiles), fp32 atomic reduction.
// ---------------------------------------------------------------------------
__global__ __launch_bounds__(256)
void k_fc_splitk(const __bf16* __restrict__ A, const __bf16* __restrict__ WT,
                 float* __restrict__ out, int kslab) {
  __shared__ __bf16 lA[2][128 * LDA];
  __shared__ __bf16 lW[2][112 * LDA];

  const int tid  = threadIdx.x;
  const int lane = tid & 31;
  const int wave = tid >> 5;
  const long kStart = (long)blockIdx.x * kslab;

  v8f acc[7] = {};
  const int mrow = wave * 16 + (lane & 15);
  const int k0   = (lane >> 4) * 8;
  const __bf16 zero = __builtin_bit_cast(__bf16, (unsigned short)0);

  // zero the N-pad rows (100..111) once, both buffers
  for (int i = tid; i < 12 * LDA; i += 256) {
    lW[0][100 * LDA + i] = zero;
    lW[1][100 * LDA + i] = zero;
  }

  auto stage = [&](int buf, long kb) {
#pragma unroll
    for (int i = 0; i < 2; ++i) {
      int idx = tid + i * 256;
      int r = idx >> 2, c = (idx & 3) * 8;
      async_b128(A + (long)r * KTOT + kb + c, &lA[buf][r * LDA + c]);
    }
    for (int idx = tid; idx < 400; idx += 256) {     // 100 rows x 4 uint4
      int r = idx >> 2, c = (idx & 3) * 8;
      async_b128(WT + (long)r * KTOT + kb + c, &lW[buf][r * LDA + c]);
    }
  };

  const int nIter = kslab >> 5;
  stage(0, kStart);
  for (int it = 0; it < nIter; ++it) {
    wait_async_all();
    __syncthreads();
    if (it + 1 < nIter) stage((it + 1) & 1, kStart + ((long)(it + 1) << 5));

    const __bf16* cA = lA[it & 1];
    const __bf16* cW = lW[it & 1];
    v16bf a = cat8(*(const v8bf*)&cA[mrow * LDA + k0],
                   *(const v8bf*)&cA[mrow * LDA + k0 + 16]);
#pragma unroll
    for (int j = 0; j < 7; ++j) {
      int n = j * 16 + (lane & 15);
      v16bf b = cat8(*(const v8bf*)&cW[n * LDA + k0],
                     *(const v8bf*)&cW[n * LDA + k0 + 16]);
      acc[j] = __builtin_amdgcn_wmma_f32_16x16x32_bf16(
          false, a, false, b, (short)0, acc[j], false, false);
    }
  }

#pragma unroll
  for (int j = 0; j < 7; ++j) {
#pragma unroll
    for (int i = 0; i < 8; ++i) {
      int m = wave * 16 + (lane >> 4) * 8 + i;
      int n = j * 16 + (lane & 15);
      if (n < NOUT) atomicAdd(&out[m * NOUT + n], acc[j][i]);
    }
  }
}

// ---------------------------------------------------------------------------
// BatchNorm over rows (mean/var per column), fused affine, bf16 out
// ---------------------------------------------------------------------------
__global__ void k_bn_stats(const float* __restrict__ x, float* __restrict__ sum,
                           float* __restrict__ sumsq, int R) {
  int c  = blockIdx.x * blockDim.x + threadIdx.x;   // gridDim.x*256 == 512
  int r0 = blockIdx.y * 256;
  int rEnd = r0 + 256; if (rEnd > R) rEnd = R;
  float s = 0.f, s2 = 0.f;
  for (int r = r0; r < rEnd; ++r) {
    float v = x[(long)r * Hh + c];
    s += v; s2 += v * v;
  }
  atomicAdd(&sum[c], s);
  atomicAdd(&sumsq[c], s2);
}

__global__ void k_bn_apply(const float* __restrict__ x, const float* __restrict__ sum,
                           const float* __restrict__ sumsq, const float* __restrict__ gamma,
                           const float* __restrict__ beta, __bf16* __restrict__ out, long R) {
  long i = (long)blockIdx.x * blockDim.x + threadIdx.x;
  if (i >= R * Hh) return;
  int c = (int)(i & (Hh - 1));
  float m = sum[c] / (float)R;
  float v = sumsq[c] / (float)R - m * m;
  out[i] = f2bf((x[i] - m) * rsqrtf(v + 1e-5f) * gamma[c] + beta[c]);
}

// ---------------------------------------------------------------------------
// Edge construction / aggregation (rel matrices are one-hot -> gathers)
// ---------------------------------------------------------------------------
__global__ void k_gather_cat2(const __bf16* __restrict__ x12, const int* __restrict__ sidx,
                              const int* __restrict__ ridx, __bf16* __restrict__ out) {
  int be = blockIdx.x; int b = be / Ee; int e = be - b * Ee;
  const uint4* s = (const uint4*)(x12 + ((long)b * Ll + sidx[e] * 2 + 0) * Hh);
  const uint4* r = (const uint4*)(x12 + ((long)b * Ll + ridx[e] * 2 + 1) * Hh);
  uint4* o = (uint4*)(out + (long)be * (2 * Hh));
  int t = threadIdx.x;                 // 128 threads, 64 uint4 per 512-bf16 segment
  if (t < 64) o[t] = s[t];
  else        o[t] = r[t - 64];
}

__global__ void k_incoming(const __bf16* __restrict__ xskip, const int* __restrict__ recList,
                           __bf16* __restrict__ out) {
  int bn = blockIdx.x; int b = bn / Nn; int n = bn - b * Nn;
  for (int c = threadIdx.x; c < Hh; c += blockDim.x) {
    float acc = 0.f;
#pragma unroll
    for (int i = 0; i < Nn - 1; ++i) {
      int e = recList[n * (Nn - 1) + i];
      acc += bf2f(xskip[((long)b * Ee + e) * Hh + c]);
    }
    out[(long)bn * Hh + c] = f2bf(acc * (1.0f / Nn));
  }
}

__global__ void k_gather_cat4(const __bf16* __restrict__ x3, const __bf16* __restrict__ xskip,
                              const int* __restrict__ sidx, const int* __restrict__ ridx,
                              __bf16* __restrict__ out) {
  int be = blockIdx.x; int b = be / Ee; int e = be - b * Ee;
  const uint4* s = (const uint4*)(x3 + ((long)b * Nn + sidx[e]) * Hh);
  const uint4* r = (const uint4*)(x3 + ((long)b * Nn + ridx[e]) * Hh);
  const uint4* k = (const uint4*)(xskip + (long)be * Hh);
  uint4* o = (uint4*)(out + (long)be * (3 * Hh));
  int t = threadIdx.x;                 // 192 threads
  if      (t < 64)  o[t] = s[t];
  else if (t < 128) o[t] = r[t - 64];
  else              o[t] = k[t - 128];
}

// ---------------------------------------------------------------------------
// Host orchestration
// ---------------------------------------------------------------------------
static inline char* carve(char*& p, size_t bytes) {
  char* r = p;
  p += (bytes + 255) & ~(size_t)255;
  return r;
}

extern "C" void kernel_launch(void* const* d_in, const int* in_sizes, int n_in,
                              void* d_out, int out_size, void* d_ws, size_t ws_size,
                              hipStream_t stream) {
  (void)in_sizes; (void)n_in; (void)out_size;
  const float* f_inputs = (const float*)d_in[0];
  const float* f_relrec = (const float*)d_in[1];
  const float* f_relsnd = (const float*)d_in[2];
  const float* f_w[4][2] = { { (const float*)d_in[3],  (const float*)d_in[5]  },
                             { (const float*)d_in[9],  (const float*)d_in[11] },
                             { (const float*)d_in[15], (const float*)d_in[17] },
                             { (const float*)d_in[21], (const float*)d_in[23] } };
  const float* f_b[4][2] = { { (const float*)d_in[4],  (const float*)d_in[6]  },
                             { (const float*)d_in[10], (const float*)d_in[12] },
                             { (const float*)d_in[16], (const float*)d_in[18] },
                             { (const float*)d_in[22], (const float*)d_in[24] } };
  const float* f_gamma[4] = { (const float*)d_in[7],  (const float*)d_in[13],
                              (const float*)d_in[19], (const float*)d_in[25] };
  const float* f_beta[4]  = { (const float*)d_in[8],  (const float*)d_in[14],
                              (const float*)d_in[20], (const float*)d_in[26] };
  const float* f_fcw = (const float*)d_in[27];
  const float* f_fcb = (const float*)d_in[28];
  float* out = (float*)d_out;

  // ---- workspace layout ----
  char* p = (char*)d_ws;
  const long wK[4][2] = { {Ff, Hh}, {2*Hh, Hh}, {Hh, Hh}, {3*Hh, Hh} };  // K per layer
  __bf16* wb[4][2];           // transposed (Hh, K) bf16
  for (int m = 0; m < 4; ++m)
    for (int l = 0; l < 2; ++l)
      wb[m][l] = (__bf16*)carve(p, (size_t)wK[m][l] * Hh * 2);
  __bf16* wb_fc  = (__bf16*)carve(p, (size_t)KTOT * NOUT * 2);   // (100, KTOT)
  __bf16* in_bf  = (__bf16*)carve(p, (size_t)Bsz * Ll * Ff * 2);
  __bf16* g1a    = (__bf16*)carve(p, (size_t)Bsz * Ll * Hh * 2);
  __bf16* x12    = (__bf16*)carve(p, (size_t)Bsz * Ll * Hh * 2);
  __bf16* cat2   = (__bf16*)carve(p, (size_t)Bsz * Ee * 2 * Hh * 2);
  __bf16* g2a    = (__bf16*)carve(p, (size_t)Bsz * Ee * Hh * 2);
  __bf16* xskip  = (__bf16*)carve(p, (size_t)Bsz * Ee * Hh * 2);
  __bf16* inc    = (__bf16*)carve(p, (size_t)Bsz * Nn * Hh * 2);
  __bf16* g3a    = (__bf16*)carve(p, (size_t)Bsz * Nn * Hh * 2);
  __bf16* x3     = (__bf16*)carve(p, (size_t)Bsz * Nn * Hh * 2);
  __bf16* cat4   = (__bf16*)carve(p, (size_t)Bsz * Ee * 3 * Hh * 2);   // also f32 temp
  __bf16* g4a    = (__bf16*)carve(p, (size_t)Bsz * Ee * Hh * 2);
  __bf16* x4     = (__bf16*)carve(p, (size_t)Bsz * Ee * Hh * 2);
  float*  stats  = (float*)carve(p, 2 * Hh * sizeof(float));
  int*    sidx   = (int*)carve(p, Ee * sizeof(int));
  int*    ridx   = (int*)carve(p, Ee * sizeof(int));
  int*    recL   = (int*)carve(p, Nn * (Nn - 1) * sizeof(int));
  if ((size_t)(p - (char*)d_ws) > ws_size) return;
  float* f32tmp = (float*)cat4;   // cat4 region is dead whenever f32tmp is live

  // ---- convert + transpose weights, convert inputs ----
  auto cvtT = [&](const float* src, __bf16* dst, long K, int N) {
    long n = K * N;
    k_cvt_transpose<<<dim3((unsigned)((n + 255) / 256)), 256, 0, stream>>>(src, dst, K, N);
  };
  for (int m = 0; m < 4; ++m)
    for (int l = 0; l < 2; ++l)
      cvtT(f_w[m][l], wb[m][l], wK[m][l], Hh);
  cvtT(f_fcw, wb_fc, KTOT, NOUT);
  {
    long n = (long)Bsz * Ll * Ff;
    k_f32_to_bf16<<<dim3((unsigned)((n + 255) / 256)), 256, 0, stream>>>(f_inputs, in_bf, n);
  }

  // ---- edge indices from one-hot rel matrices ----
  k_argmax_rows<<<dim3((Ee + 63) / 64), 64, 0, stream>>>(f_relsnd, sidx, Ee, Nn);
  k_argmax_rows<<<dim3((Ee + 63) / 64), 64, 0, stream>>>(f_relrec, ridx, Ee, Nn);
  k_build_reclist<<<1, 32, 0, stream>>>(ridx, recL);

  auto bn = [&](const float* x, int mlp, long R, __bf16* dst) {
    k_zero_f32<<<dim3(4), 256, 0, stream>>>(stats, 2 * Hh);
    k_bn_stats<<<dim3(2, (unsigned)((R + 255) / 256)), 256, 0, stream>>>(x, stats, stats + Hh, (int)R);
    k_bn_apply<<<dim3((unsigned)((R * Hh + 255) / 256)), 256, 0, stream>>>(
        x, stats, stats + Hh, f_gamma[mlp], f_beta[mlp], dst, R);
  };
  auto gemm_b = [&](const __bf16* A, int mlp, int l, void* C, long M, int K, bool outBf) {
    dim3 grid((unsigned)(M / 128), Hh / 128);
    if (outBf) k_gemm_bf16<true ><<<grid, 256, 0, stream>>>(A, wb[mlp][l], f_b[mlp][l], C, (int)M, K, Hh);
    else       k_gemm_bf16<false><<<grid, 256, 0, stream>>>(A, wb[mlp][l], f_b[mlp][l], C, (int)M, K, Hh);
  };

  const long M1 = (long)Bsz * Ll;    // 5120
  const long M2 = (long)Bsz * Ee;    // 48640
  const long M3 = (long)Bsz * Nn;    // 2560

  // mlp1: (B*L,64) -> H -> H, BN
  gemm_b(in_bf, 0, 0, g1a,    M1, Ff, true);
  gemm_b(g1a,   0, 1, f32tmp, M1, Hh, false);
  bn(f32tmp, 0, M1, x12);

  // edges: concat(senders, receivers) -> mlp2 -> BN (x_skip)
  k_gather_cat2<<<dim3((unsigned)M2), 128, 0, stream>>>(x12, sidx, ridx, cat2);
  gemm_b(cat2, 1, 0, g2a,    M2, 2 * Hh, true);
  gemm_b(g2a,  1, 1, f32tmp, M2, Hh, false);
  bn(f32tmp, 1, M2, xskip);

  // node aggregation -> mlp3 -> BN
  k_incoming<<<dim3((unsigned)M3), 256, 0, stream>>>(xskip, recL, inc);
  gemm_b(inc, 2, 0, g3a,    M3, Hh, true);
  gemm_b(g3a, 2, 1, f32tmp, M3, Hh, false);
  bn(f32tmp, 2, M3, x3);

  // concat(senders, receivers, x_skip) -> mlp4 -> BN
  k_gather_cat4<<<dim3((unsigned)M2), 192, 0, stream>>>(x3, xskip, sidx, ridx, cat4);
  gemm_b(cat4, 3, 0, g4a,    M2, 3 * Hh, true);
  gemm_b(g4a,  3, 1, f32tmp, M2, Hh, false);
  bn(f32tmp, 3, M2, x4);

  // final fc: split-K WMMA with fp32 atomic reduce (bias preset)
  k_out_init<<<dim3((Bsz * NOUT + 255) / 256), 256, 0, stream>>>(out, f_fcb);
  const int kslab = 2048;                       // 95 * 2048 == 194560
  k_fc_splitk<<<dim3(KTOT / kslab), 256, 0, stream>>>(x4, wb_fc, out, kslab);
}
